// Seq2SeqModel_566935683572
// MI455X (gfx1250) — matrix operand order repaired and checked by reading, not compiled, not run on hardware
//
#include <hip/hip_runtime.h>
#include <hip/hip_bf16.h>
#include <math.h>

typedef __bf16 bf16;
typedef __attribute__((ext_vector_type(16))) __bf16 v16bf;
typedef __attribute__((ext_vector_type(8)))  float  v8f;

#define B_  16
#define S_  64
#define T_  64
#define H_  512
#define V_  32000
#define G3  1536      // 3*H

// ---------------------------------------------------------------------------
// WMMA helpers (CDNA5 v_wmma_f32_16x16x32_bf16, wave32)
// Layouts per CDNA5 ISA §7.12.2:
//   A (16x32, row-major source, row = lane&15):
//     lanes 0-15 hold K=[k0..k0+7] and K=[k0+16..k0+23]
//     lanes 16-31 hold K=[k0+8..k0+15] and K=[k0+24..k0+31]
//   B = W^T (W row-major [N,K], col = n0+(lane&15)): same K layout as A
//     lanes 0-15 hold K=[k0..k0+7],[k0+16..k0+23]; lanes 16-31 +8
//   C/D: 8 f32/lane, element (m,n): vgpr=m%8, lane=(m/8)*16+n
// ---------------------------------------------------------------------------
__device__ __forceinline__ v8f wmma_bf16(v16bf a, v16bf b, v8f c) {
  return __builtin_amdgcn_wmma_f32_16x16x32_bf16(false, a, false, b, (short)0, c,
                                                 false, false);
}

__device__ __forceinline__ v16bf load_a_frag(const bf16* a, int lda, int k0, int lane) {
  int r  = lane & 15;
  int ko = k0 + ((lane & 16) ? 8 : 0);
  const bf16* p = a + r * lda + ko;
  v16bf f;
  ((float4*)&f)[0] = *(const float4*)(p);        // 8 bf16: K chunk 0
  ((float4*)&f)[1] = *(const float4*)(p + 16);   // 8 bf16: K chunk +16
  return f;
}

__device__ __forceinline__ v16bf load_bT_frag(const bf16* w, int ldw, int n0, int k0,
                                              int lane) {
  int c  = n0 + (lane & 15);
  int ko = k0 + ((lane & 16) ? 8 : 0);
  const bf16* p = w + c * ldw + ko;
  v16bf f;
  ((float4*)&f)[0] = *(const float4*)(p);        // 8 bf16: K chunk 0
  ((float4*)&f)[1] = *(const float4*)(p + 16);   // 8 bf16: K chunk +16
  return f;
}

__device__ __forceinline__ float sigmoidf_(float x) { return 1.0f / (1.0f + __expf(-x)); }

// ---------------------------------------------------------------------------
// Weight / embedding preparation
// ---------------------------------------------------------------------------
__global__ void k_f32_to_bf16(const float* __restrict__ src, bf16* __restrict__ dst,
                              int n) {
  for (int i = blockIdx.x * blockDim.x + threadIdx.x; i < n;
       i += gridDim.x * blockDim.x)
    dst[i] = (bf16)src[i];
}

__global__ void k_embed_enc(const int* __restrict__ x, const float* __restrict__ emb,
                            bf16* __restrict__ out) {
  int i = blockIdx.x * blockDim.x + threadIdx.x;    // over B*S*H
  if (i >= B_ * S_ * H_) return;
  int hh  = i & (H_ - 1);
  int bs  = i >> 9;
  int tok = x[bs];
  out[i] = (bf16)emb[tok * H_ + hh];
}

// ---------------------------------------------------------------------------
// Encoder: precompute input gates Gx[s][b][0:1536] = emb_x[b,s]@W_ih_e^T + b_ih_e
// grid (64, 12) x 256 threads (8 waves, 1 N-tile of 16 each)
// ---------------------------------------------------------------------------
__global__ void k_enc_gx(const bf16* __restrict__ embx, const bf16* __restrict__ Wih,
                         const float* __restrict__ bih, float* __restrict__ Gx) {
  int s = blockIdx.x;
  int w = threadIdx.x >> 5, lane = threadIdx.x & 31;
  int n0 = blockIdx.y * 128 + w * 16;
  const bf16* abase = embx + s * H_;                // row m=batch, lda = S*H
  v8f acc = {};
  for (int k0 = 0; k0 < H_; k0 += 32)
    acc = wmma_bf16(load_a_frag(abase, S_ * H_, k0, lane),
                    load_bT_frag(Wih, H_, n0, k0, lane), acc);
  int n = n0 + (lane & 15);
  float bias = bih[n];
  int mb = (lane & 16) ? 8 : 0;
  float* dst = Gx + (size_t)(s * B_) * G3;
#pragma unroll
  for (int v = 0; v < 8; ++v) dst[(mb + v) * G3 + n] = acc[v] + bias;
}

// ---------------------------------------------------------------------------
// Encoder scan: 1 block, 1024 threads (32 waves), s = 0..63 sequential.
// LDS: hbf(16K) | hfl(32K) | gh(96K) = 144K
// ---------------------------------------------------------------------------
__global__ void k_enc_scan(const float* __restrict__ Gx, const bf16* __restrict__ Whh,
                           const float* __restrict__ bhh, float* __restrict__ enc_out,
                           float* __restrict__ hout) {
  extern __shared__ char smem[];
  bf16*  hbf = (bf16*)smem;
  float* hfl = (float*)(smem + 16384);
  float* gh  = (float*)(smem + 49152);
  int tid = threadIdx.x, w = tid >> 5, lane = tid & 31;

  for (int j = tid; j < B_ * H_; j += 1024) { hfl[j] = 0.f; hbf[j] = (bf16)0.f; }
  __syncthreads();

  for (int s = 0; s < S_; ++s) {
#pragma unroll
    for (int tc = 0; tc < 3; ++tc) {                // 96 tiles over 32 waves
      int n0 = (w * 3 + tc) * 16;
      v8f acc = {};
      for (int k0 = 0; k0 < H_; k0 += 32)
        acc = wmma_bf16(load_a_frag(hbf, H_, k0, lane),
                        load_bT_frag(Whh, H_, n0, k0, lane), acc);
      int n = n0 + (lane & 15);
      float bias = bhh[n];
      int mb = (lane & 16) ? 8 : 0;
#pragma unroll
      for (int v = 0; v < 8; ++v) gh[(mb + v) * G3 + n] = acc[v] + bias;
    }
    __syncthreads();
    const float* gx = Gx + (size_t)(s * B_) * G3;
    for (int j = tid; j < B_ * H_; j += 1024) {
      int m = j >> 9, hh = j & (H_ - 1);
      const float* gm  = gx + m * G3;
      const float* ghm = gh + m * G3;
      float r  = sigmoidf_(gm[hh] + ghm[hh]);
      float z  = sigmoidf_(gm[H_ + hh] + ghm[H_ + hh]);
      float nn = tanhf(gm[2 * H_ + hh] + r * ghm[2 * H_ + hh]);
      float hv = (1.f - z) * nn + z * hfl[j];
      hfl[j] = hv;
      hbf[j] = (bf16)hv;
      enc_out[(size_t)(m * S_ + s) * H_ + hh] = hv;
    }
    __syncthreads();
  }
  for (int j = tid; j < B_ * H_; j += 1024) hout[j] = hfl[j];
}

// ---------------------------------------------------------------------------
// Decoder step (GRU + attention + W_c): 1 block, 1024 threads.
// LDS: hfl 32K | hbf 16K | ebf 16K | gi 96K | gh 96K | sc 4K  => 260K
// catbf reuses gi region, nothing reuses gh after GRU barrier.
// ---------------------------------------------------------------------------
__global__ void k_dec_step(int t, const int* __restrict__ y,
                           const float* __restrict__ emb_dec,
                           const bf16* __restrict__ Wih, const bf16* __restrict__ Whh,
                           const float* __restrict__ bih, const float* __restrict__ bhh,
                           const bf16* __restrict__ Wc, const float* __restrict__ bc,
                           const float* __restrict__ enc_out,
                           float* __restrict__ hbuf, bf16* __restrict__ dbuf) {
  extern __shared__ char smem[];
  float* hfl   = (float*)smem;                 // 32768
  bf16*  hbf   = (bf16*)(smem + 32768);        // 16384
  bf16*  ebf   = (bf16*)(smem + 49152);        // 16384
  float* gi    = (float*)(smem + 65536);       // 98304
  float* gh    = (float*)(smem + 163840);      // 98304
  float* sc    = (float*)(smem + 262144);      // 4096
  bf16*  catbf = (bf16*)(smem + 65536);        // reuse gi region (32768)
  int tid = threadIdx.x, w = tid >> 5, lane = tid & 31;

  // load h, gather decoder-input embedding
  for (int j = tid; j < B_ * H_; j += 1024) {
    float hv = hbuf[j];
    hfl[j] = hv; hbf[j] = (bf16)hv;
    int m = j >> 9, hh = j & (H_ - 1);
    int tok = y[m * T_ + t];
    ebf[j] = (bf16)emb_dec[tok * H_ + hh];
  }
  __syncthreads();

  // gi = e@W_ih^T + b_ih ; gh = h@W_hh^T + b_hh   (96 tiles each / 32 waves)
#pragma unroll
  for (int tc = 0; tc < 3; ++tc) {
    int n0 = (w * 3 + tc) * 16;
    v8f ai = {}, ah = {};
    for (int k0 = 0; k0 < H_; k0 += 32) {
      v16bf bi = load_bT_frag(Wih, H_, n0, k0, lane);
      v16bf bh = load_bT_frag(Whh, H_, n0, k0, lane);
      ai = wmma_bf16(load_a_frag(ebf, H_, k0, lane), bi, ai);
      ah = wmma_bf16(load_a_frag(hbf, H_, k0, lane), bh, ah);
    }
    int n = n0 + (lane & 15);
    float bi_ = bih[n], bh_ = bhh[n];
    int mb = (lane & 16) ? 8 : 0;
#pragma unroll
    for (int v = 0; v < 8; ++v) {
      gi[(mb + v) * G3 + n] = ai[v] + bi_;
      gh[(mb + v) * G3 + n] = ah[v] + bh_;
    }
  }
  __syncthreads();

  // GRU elementwise, persist h for next step
  for (int j = tid; j < B_ * H_; j += 1024) {
    int m = j >> 9, hh = j & (H_ - 1);
    const float* gim = gi + m * G3;
    const float* ghm = gh + m * G3;
    float r  = sigmoidf_(gim[hh] + ghm[hh]);
    float z  = sigmoidf_(gim[H_ + hh] + ghm[H_ + hh]);
    float nn = tanhf(gim[2 * H_ + hh] + r * ghm[2 * H_ + hh]);
    float hv = (1.f - z) * nn + z * hfl[j];
    hfl[j] = hv;
    hbuf[j] = hv;
  }
  __syncthreads();

  // attention scores: thread = (b,s), dot over H
  {
    int b = tid >> 6, s = tid & 63;
    const float* e  = enc_out + (size_t)(b * S_ + s) * H_;
    const float* hq = hfl + b * H_;
    float dot = 0.f;
    for (int k = 0; k < H_; ++k) dot += hq[k] * e[k];
    sc[tid] = dot;
  }
  __syncthreads();

  // softmax over s (tiny: 16 rows x 64, serial per row)
  if (tid < B_) {
    float* row = sc + tid * S_;
    float mx = row[0];
    for (int s = 1; s < S_; ++s) mx = fmaxf(mx, row[s]);
    float sum = 0.f;
    for (int s = 0; s < S_; ++s) { float e = __expf(row[s] - mx); row[s] = e; sum += e; }
    float inv = 1.f / sum;
    for (int s = 0; s < S_; ++s) row[s] *= inv;
  }
  __syncthreads();

  // attn + build cat = [h, attn] in bf16 (gi region is dead now)
  for (int j = tid; j < B_ * H_; j += 1024) {
    int b = j >> 9, hh = j & (H_ - 1);
    const float* wrow = sc + b * S_;
    const float* e    = enc_out + (size_t)b * S_ * H_ + hh;
    float a = 0.f;
    for (int s = 0; s < S_; ++s) a += wrow[s] * e[s * H_];
    catbf[b * 1024 + hh]       = (bf16)hfl[j];
    catbf[b * 1024 + H_ + hh]  = (bf16)a;
  }
  __syncthreads();

  // d = tanh(cat @ W_c^T + b_c): 32 N-tiles, one per wave, K=1024
  {
    int n0 = w * 16;
    v8f acc = {};
    for (int k0 = 0; k0 < 2 * H_; k0 += 32)
      acc = wmma_bf16(load_a_frag(catbf, 1024, k0, lane),
                      load_bT_frag(Wc, 1024, n0, k0, lane), acc);
    int n = n0 + (lane & 15);
    float bias = bc[n];
    int mb = (lane & 16) ? 8 : 0;
#pragma unroll
    for (int v = 0; v < 8; ++v)
      dbuf[(mb + v) * H_ + n] = (bf16)tanhf(acc[v] + bias);
  }
}

// ---------------------------------------------------------------------------
// logits_t = d_t @ W_out^T + b_out : M=16, N=32000 (2000 tiles), K=512.
// 250 blocks x 8 waves, one N-tile per wave. The shared A matrix d_t (16KB)
// is staged in LDS once per block: halves L2 traffic of the dominant GEMM
// (A reads drop from 32MB -> 4MB per step; B = W_out bf16 streams from L2).
// ---------------------------------------------------------------------------
__global__ void k_logits(int t, const bf16* __restrict__ dbuf,
                         const bf16* __restrict__ Wout, const float* __restrict__ bout,
                         float* __restrict__ out) {
  __shared__ bf16 albs[B_ * H_];                 // 16 KB
  int tid = threadIdx.x, w = tid >> 5, lane = tid & 31;

  // cooperative stage of A into LDS: 1024 float4s over 256 threads
  {
    const float4* src = (const float4*)dbuf;
    float4* dst = (float4*)albs;
#pragma unroll
    for (int i = 0; i < 4; ++i) dst[tid + 256 * i] = src[tid + 256 * i];
  }
  __syncthreads();

  int n0 = (blockIdx.x * 8 + w) * 16;
  v8f acc = {};
  for (int k0 = 0; k0 < H_; k0 += 32)
    acc = wmma_bf16(load_a_frag(albs, H_, k0, lane),
                    load_bT_frag(Wout, H_, n0, k0, lane), acc);
  int n = n0 + (lane & 15);
  float bias = bout[n];
  int mb = (lane & 16) ? 8 : 0;
#pragma unroll
  for (int v = 0; v < 8; ++v) {
    int m = mb + v;
    out[(size_t)(m * 63 + t) * V_ + n] = acc[v] + bias;
  }
}

// ---------------------------------------------------------------------------
// Loss: per-row logsumexp + NLL, then mean.
// ---------------------------------------------------------------------------
__global__ void k_loss_rows(const float* __restrict__ logits, const int* __restrict__ y,
                            float* __restrict__ nll) {
  __shared__ float red[256];
  int row = blockIdx.x;                 // b*63 + t
  int b = row / 63, t = row % 63;
  const float* lr = logits + (size_t)row * V_;
  int tid = threadIdx.x;
  float mx = -INFINITY;
  for (int i = tid; i < V_; i += 256) mx = fmaxf(mx, lr[i]);
  red[tid] = mx; __syncthreads();
  for (int off = 128; off > 0; off >>= 1) {
    if (tid < off) red[tid] = fmaxf(red[tid], red[tid + off]);
    __syncthreads();
  }
  mx = red[0]; __syncthreads();
  float sum = 0.f;
  for (int i = tid; i < V_; i += 256) sum += __expf(lr[i] - mx);
  red[tid] = sum; __syncthreads();
  for (int off = 128; off > 0; off >>= 1) {
    if (tid < off) red[tid] += red[tid + off];
    __syncthreads();
  }
  if (tid == 0) {
    float lse = mx + __logf(red[0]);
    int tgt = y[b * T_ + t + 1];
    nll[row] = lse - lr[tgt];
  }
}

__global__ void k_loss_final(const float* __restrict__ nll, float* __restrict__ out) {
  __shared__ float red[256];
  int tid = threadIdx.x;
  float s = 0.f;
  for (int i = tid; i < B_ * 63; i += 256) s += nll[i];
  red[tid] = s; __syncthreads();
  for (int off = 128; off > 0; off >>= 1) {
    if (tid < off) red[tid] += red[tid + off];
    __syncthreads();
  }
  if (tid == 0) out[(size_t)B_ * 63 * V_] = red[0] / (float)(B_ * 63);
}

// ---------------------------------------------------------------------------
extern "C" void kernel_launch(void* const* d_in, const int* in_sizes, int n_in,
                              void* d_out, int out_size, void* d_ws, size_t ws_size,
                              hipStream_t stream) {
  (void)in_sizes; (void)n_in; (void)out_size; (void)ws_size;
  const int*   x     = (const int*)d_in[0];
  const int*   y     = (const int*)d_in[1];
  const float* emb_e = (const float*)d_in[2];
  const float* Wih_e = (const float*)d_in[3];
  const float* Whh_e = (const float*)d_in[4];
  const float* bih_e = (const float*)d_in[5];
  const float* bhh_e = (const float*)d_in[6];
  const float* emb_d = (const float*)d_in[7];
  const float* Wih_d = (const float*)d_in[8];
  const float* Whh_d = (const float*)d_in[9];
  const float* bih_d = (const float*)d_in[10];
  const float* bhh_d = (const float*)d_in[11];
  const float* Wc    = (const float*)d_in[12];
  const float* bc    = (const float*)d_in[13];
  const float* Wout  = (const float*)d_in[14];
  const float* bout  = (const float*)d_in[15];
  float* out = (float*)d_out;

  char* ws = (char*)d_ws;
  size_t o = 0;
  auto carve = [&](size_t bytes) -> char* {
    char* p = ws + o;
    o += (bytes + 255) & ~(size_t)255;
    return p;
  };
  bf16*  wWih_e = (bf16*)carve((size_t)G3 * H_ * 2);
  bf16*  wWhh_e = (bf16*)carve((size_t)G3 * H_ * 2);
  bf16*  wWih_d = (bf16*)carve((size_t)G3 * H_ * 2);
  bf16*  wWhh_d = (bf16*)carve((size_t)G3 * H_ * 2);
  bf16*  wWc    = (bf16*)carve((size_t)H_ * 2 * H_ * 2);
  bf16*  wWout  = (bf16*)carve((size_t)V_ * H_ * 2);
  bf16*  embx   = (bf16*)carve((size_t)B_ * S_ * H_ * 2);
  float* Gx     = (float*)carve((size_t)S_ * B_ * G3 * 4);
  float* enc_o  = (float*)carve((size_t)B_ * S_ * H_ * 4);
  float* hbuf   = (float*)carve((size_t)B_ * H_ * 4);
  bf16*  dbuf   = (bf16*)carve((size_t)B_ * H_ * 2);
  float* nllb   = (float*)carve((size_t)B_ * 63 * 4);

  const int ENC_SMEM = 147456;   // 144 KB
  const int DEC_SMEM = 266240;   // 260 KB (<= 320 KB WGP LDS)
  hipFuncSetAttribute(reinterpret_cast<const void*>(&k_enc_scan),
                      hipFuncAttributeMaxDynamicSharedMemorySize, ENC_SMEM);
  hipFuncSetAttribute(reinterpret_cast<const void*>(&k_dec_step),
                      hipFuncAttributeMaxDynamicSharedMemorySize, DEC_SMEM);

  // weights -> bf16 once per launch (streams ~120MB at HBM rate: negligible)
  k_f32_to_bf16<<<2048, 256, 0, stream>>>(Wih_e, wWih_e, G3 * H_);
  k_f32_to_bf16<<<2048, 256, 0, stream>>>(Whh_e, wWhh_e, G3 * H_);
  k_f32_to_bf16<<<2048, 256, 0, stream>>>(Wih_d, wWih_d, G3 * H_);
  k_f32_to_bf16<<<2048, 256, 0, stream>>>(Whh_d, wWhh_d, G3 * H_);
  k_f32_to_bf16<<<2048, 256, 0, stream>>>(Wc, wWc, H_ * 2 * H_);
  k_f32_to_bf16<<<4096, 256, 0, stream>>>(Wout, wWout, V_ * H_);

  // encoder
  k_embed_enc<<<(B_ * S_ * H_ + 255) / 256, 256, 0, stream>>>(x, emb_e, embx);
  dim3 gxg(S_, 12);
  k_enc_gx<<<gxg, 256, 0, stream>>>(embx, wWih_e, bih_e, Gx);
  k_enc_scan<<<1, 1024, ENC_SMEM, stream>>>(Gx, wWhh_e, bhh_e, enc_o, hbuf);

  // decoder: sequential chain through h; logits GEMM fans out per step
  for (int t = 0; t < 63; ++t) {
    k_dec_step<<<1, 1024, DEC_SMEM, stream>>>(t, y, emb_d, wWih_d, wWhh_d, bih_d,
                                              bhh_d, wWc, bc, enc_o, hbuf, dbuf);
    k_logits<<<250, 256, 0, stream>>>(t, dbuf, wWout, bout, out);
  }

  // loss
  k_loss_rows<<<B_ * 63, 256, 0, stream>>>(out, y, nllb);
  k_loss_final<<<1, 256, 0, stream>>>(nllb, out);
}
